// Memory_6571299963345
// MI455X (gfx1250) — matrix operand order repaired and checked by reading, not compiled, not run on hardware
//
#include <hip/hip_runtime.h>
#include <hip/hip_bf16.h>
#include <math.h>

// ---------------------------------------------------------------------------
// MI455X (gfx1250) implementation. wave32, WMMA 16x16 tiles.
//   bf16 WMMA  : tall activation GEMMs (N = 16384 rows)
//   f32  WMMA  : memory GEMM (537MB Wr1 stays fp32), qkv, rel0
//   TDM        : async tensor_load_to_lds double-buffering of Wr1 B-tiles
// ---------------------------------------------------------------------------

typedef __attribute__((ext_vector_type(16))) __bf16 v16bf;
typedef __attribute__((ext_vector_type(8)))  float  v8f;
typedef __attribute__((ext_vector_type(2)))  float  v2f;
typedef __attribute__((ext_vector_type(4)))  unsigned int u32x4;
typedef __attribute__((ext_vector_type(8)))  unsigned int u32x8;

#define NROWS 16384   // B*H*W
#define CDIM  512
#define MMEM  100
#define QKV3  300

// LDS offset (bytes) of a __shared__ object: generic -> AS(3) -> integer.
__device__ __forceinline__ unsigned lds_offset_of(const void* p) {
  return (unsigned)(size_t)(const __attribute__((address_space(3))) char*)p;
}

// ---------------- elementwise / small kernels ----------------

__global__ void k_f2bf(const float* __restrict__ s, __bf16* __restrict__ d, int n) {
  int i = blockIdx.x * 256 + threadIdx.x;
  if (i < n) d[i] = (__bf16)s[i];
}

__global__ void k_zero(float* p, int n) {
  int i = blockIdx.x * 256 + threadIdx.x;
  if (i < n) p[i] = 0.f;
}

__global__ void k_initmem(float* mem, const float* __restrict__ br1) {
  int i = blockIdx.x * 256 + threadIdx.x;
  if (i < MMEM * CDIM) mem[i] = br1[i & 511];
}

// fea (4,512,64,64) -> row-major qn (16384,512) bf16, L2-normalized rows.
__global__ void k_qnorm(const float* __restrict__ fea, __bf16* __restrict__ qn) {
  int bh = blockIdx.x;                 // 0..255
  int b = bh >> 6, h = bh & 63;
  int w = threadIdx.x & 63, cc = threadIdx.x >> 6;
  __shared__ float part[4][64];
  __shared__ float inv[64];
  const float* base = fea + (size_t)b * CDIM * 4096 + h * 64 + w;
  float ss = 0.f;
  for (int c = cc; c < CDIM; c += 4) {
    float v = base[(size_t)c * 4096];
    ss += v * v;
  }
  part[cc][w] = ss;
  __syncthreads();
  if (cc == 0) {
    float t = part[0][w] + part[1][w] + part[2][w] + part[3][w];
    inv[w] = 1.0f / fmaxf(sqrtf(t), 1e-12f);
  }
  __syncthreads();
  int n = (b * 64 + h) * 64 + w;
  float s = inv[w];
  __bf16* row = qn + (size_t)n * CDIM;
  for (int c = cc; c < CDIM; c += 4) row[c] = (__bf16)(base[(size_t)c * 4096] * s);
}

// softmax over 100 cols, one block per row, writes bf16
__global__ void k_softmax100(const float* __restrict__ in, __bf16* __restrict__ out) {
  int row = blockIdx.x, t = threadIdx.x;         // 128 threads
  __shared__ float red[128];
  float x = (t < MMEM) ? in[(size_t)row * MMEM + t] : -3.4e38f;
  red[t] = x; __syncthreads();
  for (int s = 64; s > 0; s >>= 1) { if (t < s) red[t] = fmaxf(red[t], red[t + s]); __syncthreads(); }
  float mx = red[0]; __syncthreads();
  float e = (t < MMEM) ? __expf(x - mx) : 0.f;
  red[t] = e; __syncthreads();
  for (int s = 64; s > 0; s >>= 1) { if (t < s) red[t] += red[t + s]; __syncthreads(); }
  float inv = 1.0f / red[0];
  if (t < MMEM) out[(size_t)row * MMEM + t] = (__bf16)(e * inv);
}

// s4[f] = sum_{n,d} U[n,d] * rel_w[n,d,f]   (grid: (2, 100))
__global__ void k_s4(const float* __restrict__ U, const float* __restrict__ rel_w,
                     float* __restrict__ s4) {
  int f = blockIdx.x * 256 + threadIdx.x;
  int n = blockIdx.y;
  const float* rw = rel_w + (size_t)n * CDIM * CDIM + f;
  const float* u  = U + (size_t)n * CDIM;
  float acc = 0.f;
  for (int d = 0; d < CDIM; ++d) acc += u[d] * rw[(size_t)d * CDIM];
  atomicAdd(&s4[f], acc);
}

// s2[d] = sum_n ip2[n,d]   (grid: (2, 64), 256-row chunks)
__global__ void k_s2(const __bf16* __restrict__ ip2, float* __restrict__ s2) {
  int d = blockIdx.x * 256 + threadIdx.x;
  size_t n0 = (size_t)blockIdx.y * 256;
  float acc = 0.f;
  for (int r = 0; r < 256; ++r) acc += (float)ip2[(n0 + r) * CDIM + d];
  atomicAdd(&s2[d], acc);
}

// item = outer(s4,s2) + item_w + G
__global__ void k_item(const float* __restrict__ s4, const float* __restrict__ s2,
                       const float* __restrict__ item_w, const float* __restrict__ G,
                       float* __restrict__ item) {
  int i = blockIdx.x * 256 + threadIdx.x;
  if (i < CDIM * CDIM) {
    int d = i >> 9, f = i & 511;
    item[i] = s4[d] * s2[f] + item_w[i] + G[i];
  }
}

// LayerNorm(eps=512.0 quirk) over 300 cols; writes qT/kT/vT (each 100x512)
__global__ void k_ln_qkvT(const float* __restrict__ qkv, const float* __restrict__ g,
                          const float* __restrict__ bb, float* __restrict__ qT,
                          float* __restrict__ kT, float* __restrict__ vT) {
  int row = blockIdx.x, t = threadIdx.x;         // 512 threads
  __shared__ float red[512];
  float x = (t < QKV3) ? qkv[(size_t)row * QKV3 + t] : 0.f;
  red[t] = x; __syncthreads();
  for (int s = 256; s > 0; s >>= 1) { if (t < s) red[t] += red[t + s]; __syncthreads(); }
  float mean = red[0] * (1.0f / 300.0f);
  __syncthreads();
  float dd = (t < QKV3) ? (x - mean) : 0.f;
  red[t] = dd * dd; __syncthreads();
  for (int s = 256; s > 0; s >>= 1) { if (t < s) red[t] += red[t + s]; __syncthreads(); }
  float var = red[0] * (1.0f / 300.0f);
  if (t < QKV3) {
    float y = dd * rsqrtf(var + 512.0f) * g[t] + bb[t];
    if (t < 100)      qT[(size_t)t * CDIM + row] = y;
    else if (t < 200) kT[(size_t)(t - 100) * CDIM + row] = y;
    else              vT[(size_t)(t - 200) * CDIM + row] = y;
  }
}

// l2norm rows of memory (100x512) -> mn bf16 (row major) + mnT bf16 (512x100)
__global__ void k_mnorm(const float* __restrict__ mem, __bf16* __restrict__ mn_bf,
                        __bf16* __restrict__ mnT_bf) {
  int row = blockIdx.x, t = threadIdx.x;         // 256 threads
  __shared__ float red[256];
  float ss = 0.f;
  for (int c = t; c < CDIM; c += 256) { float v = mem[(size_t)row * CDIM + c]; ss += v * v; }
  red[t] = ss; __syncthreads();
  for (int s = 128; s > 0; s >>= 1) { if (t < s) red[t] += red[t + s]; __syncthreads(); }
  float inv = 1.0f / fmaxf(sqrtf(red[0]), 1e-12f);
  for (int c = t; c < CDIM; c += 256) {
    float v = mem[(size_t)row * CDIM + c] * inv;
    mn_bf[(size_t)row * CDIM + c] = (__bf16)v;
    mnT_bf[(size_t)c * MMEM + row] = (__bf16)v;
  }
}

// out (4,1024,64,64): channels [0,512)=fea, [512,1024)=mf (row-major N x 512)
__global__ void k_concat(const float* __restrict__ fea, const float* __restrict__ mf,
                         float* __restrict__ out) {
  int bc = blockIdx.x;                // 4096 = 4 * 1024
  int b = bc >> 10, c = bc & 1023;
  float* o = out + ((size_t)b * 1024 + c) * 4096;
  if (c < CDIM) {
    const float* src = fea + ((size_t)b * CDIM + c) * 4096;
    for (int p = threadIdx.x; p < 4096; p += 256) o[p] = src[p];
  } else {
    int cm = c - CDIM;
    for (int p = threadIdx.x; p < 4096; p += 256)
      o[p] = mf[((size_t)(b * 4096 + p)) * CDIM + cm];
  }
}

// ---------------- bf16 WMMA GEMM (f32 accumulate) ----------------
template <bool A_TRANS, bool OUT_BF16, bool ATOMIC>
__global__ void k_gemm_bf16(const __bf16* __restrict__ A, const __bf16* __restrict__ Bm,
                            const float* __restrict__ bias, float* __restrict__ Cf,
                            __bf16* __restrict__ Cb, int Mdim, int Ndim, int Kdim,
                            int lda, int ldb, int ldc, int kchunk) {
  __shared__ __bf16 As[64][40];
  __shared__ __bf16 Bs[32][72];
  const int tid = threadIdx.x;
  const int lane = tid & 31, wv = tid >> 5;
  const int l15 = lane & 15, half = lane >> 4;
  const int m0 = blockIdx.y * 64, n0 = blockIdx.x * 64;
  int kb = blockIdx.z * kchunk;
  int ke = kb + kchunk; if (ke > Kdim) ke = Kdim;

  v8f acc[4];
  for (int i = 0; i < 4; ++i)
    for (int e = 0; e < 8; ++e) acc[i][e] = 0.f;

  for (int k0 = kb; k0 < ke; k0 += 32) {
    for (int id = tid; id < 64 * 32; id += 128) {
      int r, c;
      if (A_TRANS) { c = id >> 6; r = id & 63; }   // A stored (K x M)
      else         { r = id >> 5; c = id & 31; }   // A stored (M x K)
      int gm = m0 + r, gk = k0 + c;
      __bf16 v = (__bf16)0.f;
      if (gm < Mdim && gk < ke)
        v = A_TRANS ? A[(size_t)gk * lda + gm] : A[(size_t)gm * lda + gk];
      As[r][c] = v;
    }
    for (int id = tid; id < 32 * 64; id += 128) {
      int r = id >> 6, c = id & 63;
      int gk = k0 + r, gn = n0 + c;
      Bs[r][c] = (gk < ke && gn < Ndim) ? Bm[(size_t)gk * ldb + gn] : (__bf16)0.f;
    }
    __syncthreads();

    v16bf a, bf;
#pragma unroll
    for (int e = 0; e < 16; ++e)
      a[e] = As[wv * 16 + l15][((e < 8) ? e : e + 8) + half * 8];
#pragma unroll
    for (int ct = 0; ct < 4; ++ct) {
#pragma unroll
      for (int e = 0; e < 16; ++e)
        bf[e] = Bs[((e < 8) ? e : e + 8) + half * 8][ct * 16 + l15];
      acc[ct] = __builtin_amdgcn_wmma_f32_16x16x32_bf16(
          false, a, false, bf, (short)0, acc[ct], false, false);
    }
    __syncthreads();
  }

#pragma unroll
  for (int ct = 0; ct < 4; ++ct) {
    int gn = n0 + ct * 16 + l15;
    if (gn >= Ndim) continue;
    float bv = bias ? bias[gn] : 0.f;
#pragma unroll
    for (int e = 0; e < 8; ++e) {
      int gm = m0 + wv * 16 + e + 8 * half;
      if (gm >= Mdim) continue;
      float v = acc[ct][e] + bv;
      size_t idx = (size_t)gm * ldc + gn;
      if (ATOMIC)            atomicAdd(&Cf[idx], v);
      else if (OUT_BF16)     Cb[idx] = (__bf16)v;
      else                   Cf[idx] = v;
    }
  }
}

// ---------------- f32 WMMA GEMM (v_wmma_f32_16x16x4_f32) ----------------
template <bool ATOMIC>
__global__ void k_gemm_f32(const float* __restrict__ A, const float* __restrict__ Bm,
                           const float* __restrict__ bias, float* __restrict__ Cf,
                           int Mdim, int Ndim, int Kdim,
                           int lda, int ldb, int ldc, int kchunk) {
  __shared__ float As[64][20];
  __shared__ float Bs[16][68];
  const int tid = threadIdx.x;
  const int lane = tid & 31, wv = tid >> 5;
  const int l15 = lane & 15, half = lane >> 4;
  const int m0 = blockIdx.y * 64, n0 = blockIdx.x * 64;
  int kb = blockIdx.z * kchunk;
  int ke = kb + kchunk; if (ke > Kdim) ke = Kdim;

  v8f acc[4];
  for (int i = 0; i < 4; ++i)
    for (int e = 0; e < 8; ++e) acc[i][e] = 0.f;

  for (int k0 = kb; k0 < ke; k0 += 16) {
    for (int id = tid; id < 64 * 16; id += 128) {
      int r = id >> 4, c = id & 15;
      int gm = m0 + r, gk = k0 + c;
      As[r][c] = (gm < Mdim && gk < ke) ? A[(size_t)gm * lda + gk] : 0.f;
    }
    for (int id = tid; id < 16 * 64; id += 128) {
      int r = id >> 6, c = id & 63;
      int gk = k0 + r, gn = n0 + c;
      Bs[r][c] = (gk < ke && gn < Ndim) ? Bm[(size_t)gk * ldb + gn] : 0.f;
    }
    if (k0 + 16 < ke)
      __builtin_prefetch(&Bm[(size_t)(k0 + 16) * ldb + n0 + (tid & 63)], 0, 1);
    __syncthreads();

#pragma unroll
    for (int ks = 0; ks < 4; ++ks) {
      int kk = ks * 4;
      v2f a;
      a[0] = As[wv * 16 + l15][kk + 0 + 2 * half];
      a[1] = As[wv * 16 + l15][kk + 1 + 2 * half];
#pragma unroll
      for (int ct = 0; ct < 4; ++ct) {
        v2f b;
        b[0] = Bs[kk + 0 + 2 * half][ct * 16 + l15];
        b[1] = Bs[kk + 1 + 2 * half][ct * 16 + l15];
        acc[ct] = __builtin_amdgcn_wmma_f32_16x16x4_f32(
            false, a, false, b, (short)0, acc[ct], false, false);
      }
    }
    __syncthreads();
  }

#pragma unroll
  for (int ct = 0; ct < 4; ++ct) {
    int gn = n0 + ct * 16 + l15;
    if (gn >= Ndim) continue;
    float bv = bias ? bias[gn] : 0.f;
#pragma unroll
    for (int e = 0; e < 8; ++e) {
      int gm = m0 + wv * 16 + e + 8 * half;
      if (gm >= Mdim) continue;
      float v = acc[ct][e] + bv;
      size_t idx = (size_t)gm * ldc + gn;
      if (ATOMIC) atomicAdd(&Cf[idx], v);
      else        Cf[idx] = v;
    }
  }
}

// ---------- memory GEMM with TDM double-buffered B staging ------------------
// memory(100x512) += rel_w_new(100x262144) @ Wr1(262144x512), split-K grid.z.
// B tiles (16x64 f32) are moved global->LDS by the Tensor Data Mover
// (tensor_load_to_lds, TENSORcnt), issued by wave 0, double buffered.
__global__ void k_gemm_wr1_tdm(const float* __restrict__ A, const float* __restrict__ Bm,
                               float* __restrict__ Cf) {
  __shared__ float As[64][20];
  __shared__ float Bs[2][16][64];
  const int tid = threadIdx.x;                       // 128 threads = 4 waves
  const int lane = tid & 31, wv = tid >> 5;
  const int l15 = lane & 15, half = lane >> 4;
  const int m0 = blockIdx.y * 64, n0 = blockIdx.x * 64;
  const int kb = blockIdx.z * 4096, ke = kb + 4096;  // K = 262144 exactly

  v8f acc[4];
  for (int i = 0; i < 4; ++i)
    for (int e = 0; e < 8; ++e) acc[i][e] = 0.f;

  const unsigned lds_b0 = lds_offset_of(&Bs[0][0][0]);

  // Issue one TDM 2D-tile load: 16 rows x 64 cols f32 from the 512-wide Wr1.
  auto tdm_issue = [&](int buf, int k0) {
    unsigned long long ga =
        (unsigned long long)(const void*)(Bm + (size_t)k0 * CDIM + n0);
    u32x4 g0;
    g0[0] = 1u;                                     // count=1, user descriptor
    g0[1] = lds_b0 + (unsigned)buf * 16u * 64u * 4u;// lds_addr (bytes)
    g0[2] = (unsigned)ga;                           // global_addr[31:0]
    g0[3] = (unsigned)((ga >> 32) & 0x1FFFFFFu)     // global_addr[56:32]
            | (2u << 30);                           // type = 2 ("image")
    u32x8 g1;
    g1[0] = 2u << 16;          // workgroup_mask=0, data_size=2 (4 bytes)
    g1[1] = 512u << 16;        // tensor_dim0 = 512 (low 16 bits)
    g1[2] = 0u;                // tensor_dim0 hi, tensor_dim1 lo16 (dim1=65536)
    g1[3] = 1u | (64u << 16);  // tensor_dim1 hi16 = 1, tile_dim0 = 64
    g1[4] = 16u;               // tile_dim1 = 16, tile_dim2 = 0
    g1[5] = 512u;              // tensor_dim0_stride = 512 (low 32)
    g1[6] = 0u;                // stride hi, tensor_dim1_stride lo (unused, 2D)
    g1[7] = 0u;
    asm volatile("tensor_load_to_lds %0, %1" :: "s"(g0), "s"(g1) : "memory");
  };

  if (tid < 32) tdm_issue(0, kb);                    // preload buffer 0

  int it = 0;
  for (int k0 = kb; k0 < ke; k0 += 16, ++it) {
    int cur = it & 1;
    __syncthreads();   // previous iteration's reads of As / Bs[cur^1] done

    // stage A tile (normal VMEM path; only 100 rows total, reused 8x from L2)
    for (int id = tid; id < 64 * 16; id += 128) {
      int r = id >> 4, c = id & 15;
      int gm = m0 + r;
      As[r][c] = (gm < MMEM) ? A[(size_t)gm * (CDIM * CDIM) + k0 + c] : 0.f;
    }

    if (tid < 32) {    // wave 0 drives the TDM pipeline
      if (k0 + 16 < ke) {
        tdm_issue(cur ^ 1, k0 + 16);
        __builtin_amdgcn_s_wait_tensorcnt(1);   // current tile complete
      } else {
        __builtin_amdgcn_s_wait_tensorcnt(0);
      }
    }
    __syncthreads();   // As staged + Bs[cur] landed

#pragma unroll
    for (int ks = 0; ks < 4; ++ks) {
      int kk = ks * 4;
      v2f a;
      a[0] = As[wv * 16 + l15][kk + 0 + 2 * half];
      a[1] = As[wv * 16 + l15][kk + 1 + 2 * half];
#pragma unroll
      for (int ct = 0; ct < 4; ++ct) {
        v2f b;
        b[0] = Bs[cur][kk + 0 + 2 * half][ct * 16 + l15];
        b[1] = Bs[cur][kk + 1 + 2 * half][ct * 16 + l15];
        acc[ct] = __builtin_amdgcn_wmma_f32_16x16x4_f32(
            false, a, false, b, (short)0, acc[ct], false, false);
      }
    }
  }

#pragma unroll
  for (int ct = 0; ct < 4; ++ct) {
    int gn = n0 + ct * 16 + l15;
#pragma unroll
    for (int e = 0; e < 8; ++e) {
      int gm = m0 + wv * 16 + e + 8 * half;
      if (gm < MMEM) atomicAdd(&Cf[(size_t)gm * CDIM + gn], acc[ct][e]);
    }
  }
}

// ---------------- rel0: rel_w_new[i,d,f] = rel_w + tanh(q_d x k_d) @ V ------
#define REL0_LDS_BYTES ((112 * 113 + 112 * 129 + 224) * 4)
__global__ void k_rel0(const float* __restrict__ qT, const float* __restrict__ kT,
                       const float* __restrict__ vT, const float* __restrict__ rel_w,
                       float* __restrict__ relw_new) {
  extern __shared__ char smem[];
  float* Td = (float*)smem;           // 112 x 113
  float* Vs = Td + 112 * 113;         // 112 x 129
  float* qd = Vs + 112 * 129;         // 112
  float* kd = qd + 112;               // 112
  int d = blockIdx.y, f0 = blockIdx.x * 128;
  int tid = threadIdx.x;
  if (tid < 112) {
    qd[tid] = (tid < MMEM) ? qT[(size_t)tid * CDIM + d] : 0.f;
    kd[tid] = (tid < MMEM) ? kT[(size_t)tid * CDIM + d] : 0.f;
  }
  __syncthreads();
  for (int id = tid; id < 112 * 112; id += 256) {
    int i = id / 112, j = id % 112;
    Td[i * 113 + j] = (i < MMEM && j < MMEM) ? tanhf(qd[i] * kd[j]) : 0.f;
  }
  for (int id = tid; id < 112 * 128; id += 256) {
    int j = id >> 7, f = id & 127;
    Vs[j * 129 + f] = (j < MMEM) ? vT[(size_t)j * CDIM + f0 + f] : 0.f;
  }
  __syncthreads();

  int lane = tid & 31, wv = tid >> 5;
  int l15 = lane & 15, half = lane >> 4;
  v8f acc[7];
  for (int s = 0; s < 7; ++s)
    for (int e = 0; e < 8; ++e) acc[s][e] = 0.f;

  for (int k0 = 0; k0 < 112; k0 += 4) {
#pragma unroll
    for (int s = 0; s < 7; ++s) {
      int t = wv * 7 + s;               // 56 tiles: 7 row-tiles x 8 col-tiles
      int rt = t >> 3, ct = t & 7;
      v2f a, b;
      a[0] = Td[(rt * 16 + l15) * 113 + k0 + 0 + 2 * half];
      a[1] = Td[(rt * 16 + l15) * 113 + k0 + 1 + 2 * half];
      b[0] = Vs[(k0 + 0 + 2 * half) * 129 + ct * 16 + l15];
      b[1] = Vs[(k0 + 1 + 2 * half) * 129 + ct * 16 + l15];
      acc[s] = __builtin_amdgcn_wmma_f32_16x16x4_f32(
          false, a, false, b, (short)0, acc[s], false, false);
    }
  }

#pragma unroll
  for (int s = 0; s < 7; ++s) {
    int t = wv * 7 + s;
    int rt = t >> 3, ct = t & 7;
    int f = f0 + ct * 16 + l15;
#pragma unroll
    for (int e = 0; e < 8; ++e) {
      int i = rt * 16 + e + 8 * half;
      if (i < MMEM) {
        size_t idx = ((size_t)i * CDIM + d) * CDIM + f;
        relw_new[idx] = rel_w[idx] + acc[s][e];
      }
    }
  }
}

// ---------------------------------------------------------------------------

extern "C" void kernel_launch(void* const* d_in, const int* in_sizes, int n_in,
                              void* d_out, int out_size, void* d_ws, size_t ws_size,
                              hipStream_t stream) {
  (void)in_sizes; (void)n_in; (void)out_size; (void)ws_size;
  const float* fea    = (const float*)d_in[0];
  const float* item_w = (const float*)d_in[1];
  const float* rel_w  = (const float*)d_in[2];
  const float* W1 = (const float*)d_in[3];   const float* b1 = (const float*)d_in[4];
  const float* W2 = (const float*)d_in[5];   const float* b2 = (const float*)d_in[6];
  const float* W3 = (const float*)d_in[7];   const float* b3 = (const float*)d_in[8];
  const float* Wqkv = (const float*)d_in[9]; const float* bqkv = (const float*)d_in[10];
  const float* ln_g = (const float*)d_in[11]; const float* ln_b = (const float*)d_in[12];
  const float* Wr1 = (const float*)d_in[13]; const float* br1 = (const float*)d_in[14];
  // d_in[15]=Wr2, d_in[16]=br2 are dead in the reference (faithful quirk).

  char* w = (char*)d_ws;
  size_t off = 0;
  auto carve = [&](size_t bytes) -> void* {
    void* p = w + off;
    off += (bytes + 255) & ~(size_t)255;
    return p;
  };

  __bf16* qn_bf   = (__bf16*)carve((size_t)NROWS * CDIM * 2);
  __bf16* w1b     = (__bf16*)carve((size_t)CDIM * CDIM * 2);
  __bf16* w2b     = (__bf16*)carve((size_t)CDIM * CDIM * 2);
  __bf16* w3b     = (__bf16*)carve((size_t)CDIM * MMEM * 2);
  __bf16* ip1b    = (__bf16*)carve((size_t)NROWS * CDIM * 2);
  __bf16* ip2b    = (__bf16*)carve((size_t)NROWS * CDIM * 2);
  float*  logits  = (float*)carve((size_t)NROWS * MMEM * 4);   // reused twice
  __bf16* prob_bf = (__bf16*)carve((size_t)NROWS * MMEM * 2);  // ip3, then addr
  float*  U       = (float*)carve((size_t)MMEM * CDIM * 4);
  float*  s4      = (float*)carve(CDIM * 4);
  float*  s2      = (float*)carve(CDIM * 4);
  float*  G       = (float*)carve((size_t)CDIM * CDIM * 4);
  float*  item    = (float*)carve((size_t)CDIM * CDIM * 4);
  float*  qkv     = (float*)carve((size_t)CDIM * QKV3 * 4);
  float*  qT      = (float*)carve((size_t)MMEM * CDIM * 4);
  float*  kT      = (float*)carve((size_t)MMEM * CDIM * 4);
  float*  vT      = (float*)carve((size_t)MMEM * CDIM * 4);
  float*  relw_nw = (float*)carve((size_t)MMEM * CDIM * CDIM * 4);  // 100 MiB
  float*  memv    = (float*)carve((size_t)MMEM * CDIM * 4);
  __bf16* mn_bf   = (__bf16*)carve((size_t)MMEM * CDIM * 2);
  __bf16* mnT_bf  = (__bf16*)carve((size_t)CDIM * MMEM * 2);
  float*  mf      = (float*)carve((size_t)NROWS * CDIM * 4);

  // weight conversions (small)
  k_f2bf<<<dim3((CDIM * CDIM + 255) / 256), 256, 0, stream>>>(W1, w1b, CDIM * CDIM);
  k_f2bf<<<dim3((CDIM * CDIM + 255) / 256), 256, 0, stream>>>(W2, w2b, CDIM * CDIM);
  k_f2bf<<<dim3((CDIM * MMEM + 255) / 256), 256, 0, stream>>>(W3, w3b, CDIM * MMEM);

  // qn = l2norm(transpose(fea))
  k_qnorm<<<dim3(256), 256, 0, stream>>>(fea, qn_bf);

  // ip1/ip2 (bf16 out), ip3 logits (f32)
  k_gemm_bf16<false, true, false><<<dim3(8, 256, 1), 128, 0, stream>>>(
      qn_bf, w1b, b1, nullptr, ip1b, NROWS, CDIM, CDIM, CDIM, CDIM, CDIM, CDIM);
  k_gemm_bf16<false, true, false><<<dim3(8, 256, 1), 128, 0, stream>>>(
      qn_bf, w2b, b2, nullptr, ip2b, NROWS, CDIM, CDIM, CDIM, CDIM, CDIM, CDIM);
  k_gemm_bf16<false, false, false><<<dim3(2, 256, 1), 128, 0, stream>>>(
      qn_bf, w3b, b3, logits, nullptr, NROWS, MMEM, CDIM, CDIM, MMEM, MMEM, CDIM);
  k_softmax100<<<dim3(NROWS), 128, 0, stream>>>(logits, prob_bf);

  // zero split-K accumulators
  k_zero<<<dim3((MMEM * CDIM + 255) / 256), 256, 0, stream>>>(U, MMEM * CDIM);
  k_zero<<<dim3((CDIM * CDIM + 255) / 256), 256, 0, stream>>>(G, CDIM * CDIM);
  k_zero<<<dim3(2), 256, 0, stream>>>(s4, CDIM);
  k_zero<<<dim3(2), 256, 0, stream>>>(s2, CDIM);

  // U = ip3^T @ ip2  (split-K over N=16384)
  k_gemm_bf16<true, false, true><<<dim3(8, 2, 16), 128, 0, stream>>>(
      prob_bf, ip2b, nullptr, U, nullptr, MMEM, CDIM, NROWS, MMEM, CDIM, CDIM, 1024);
  k_s4<<<dim3(2, MMEM), 256, 0, stream>>>(U, rel_w, s4);
  k_s2<<<dim3(2, 64), 256, 0, stream>>>(ip2b, s2);

  // G = ip1^T @ ip1 (split-K)
  k_gemm_bf16<true, false, true><<<dim3(8, 8, 8), 128, 0, stream>>>(
      ip1b, ip1b, nullptr, G, nullptr, CDIM, CDIM, NROWS, CDIM, CDIM, CDIM, 2048);

  // item = outer(s4, s2) + item_w + G
  k_item<<<dim3((CDIM * CDIM + 255) / 256), 256, 0, stream>>>(s4, s2, item_w, G, item);

  // qkv = item @ Wqkv + bqkv  (f32 WMMA)
  k_gemm_f32<false><<<dim3(5, 8, 1), 128, 0, stream>>>(
      item, Wqkv, bqkv, qkv, CDIM, QKV3, CDIM, CDIM, QKV3, QKV3, CDIM);

  // LayerNorm (eps = 512.0) + transpose into q/k/v (100 x 512)
  k_ln_qkvT<<<dim3(CDIM), 512, 0, stream>>>(qkv, ln_g, ln_b, qT, kT, vT);

  // rel_w_new = rel_w + rel0 (512 per-d f32 WMMA GEMMs)
  k_rel0<<<dim3(4, CDIM), 256, REL0_LDS_BYTES, stream>>>(qT, kT, vT, rel_w, relw_nw);

  // memory = rel_w_new @ Wr1 + br1  (f32 WMMA + TDM B staging, split-K)
  k_initmem<<<dim3((MMEM * CDIM + 255) / 256), 256, 0, stream>>>(memv, br1);
  k_gemm_wr1_tdm<<<dim3(8, 2, 64), 128, 0, stream>>>(relw_nw, Wr1, memv);

  // mn = l2norm(memory); addr = softmax(qn @ mn^T); mf = addr @ mn
  k_mnorm<<<dim3(MMEM), 256, 0, stream>>>(memv, mn_bf, mnT_bf);
  k_gemm_bf16<false, false, false><<<dim3(2, 256, 1), 128, 0, stream>>>(
      qn_bf, mnT_bf, nullptr, logits, nullptr, NROWS, MMEM, CDIM, CDIM, MMEM, MMEM, CDIM);
  k_softmax100<<<dim3(NROWS), 128, 0, stream>>>(logits, prob_bf);
  k_gemm_bf16<false, false, false><<<dim3(8, 256, 1), 128, 0, stream>>>(
      prob_bf, mn_bf, nullptr, mf, nullptr, NROWS, CDIM, MMEM, MMEM, CDIM, CDIM, MMEM);

  // out = concat([fea, mf(bchw)], axis=1)
  k_concat<<<dim3(4096), 256, 0, stream>>>(fea, mf, (float*)d_out);
}
